// SuperWindowAttention_71975061947032
// MI455X (gfx1250) — matrix operand order repaired and checked by reading, not compiled, not run on hardware
//
#include <hip/hip_runtime.h>
#include <hip/hip_bf16.h>

typedef __attribute__((ext_vector_type(16))) _Float16 v16h;
typedef __attribute__((ext_vector_type(8)))  _Float16 v8h;
typedef __attribute__((ext_vector_type(4)))  _Float16 v4h;
typedef __attribute__((ext_vector_type(8)))  float    v8f;

#define B_TOT   4096
#define NTOK    49
#define HEADS   12
#define HD      32
#define DIM     384
#define QKD     384
#define NW      64
#define MTOT    (B_TOT * NTOK)          /* 200704 = 3136 * 64, exact tiles */
#define KSTEPS  (DIM / 32)              /* 12 */
#define SCALE   0.17677669529663687f    /* 32^-0.5 */

/* CDNA5 async global->LDS copy (GLOBAL_LOAD_ASYNC_TO_LDS_B128, ASYNCcnt). */
#define USE_ASYNC_COPY 1

__device__ __forceinline__ void async_copy_b128(const void* gptr, void* lptr) {
#if USE_ASYNC_COPY
  unsigned lds_off = (unsigned)(size_t)lptr;   /* low 32b of generic = LDS offset */
  asm volatile("global_load_async_to_lds_b128 %0, %1, off"
               :: "v"(lds_off), "v"(gptr) : "memory");
#else
  *(v8h*)lptr = *(const v8h*)gptr;
#endif
}

__device__ __forceinline__ void async_wait() {
#if USE_ASYNC_COPY
  asm volatile("s_wait_asynccnt 0x0" ::: "memory");
#endif
}

/* ---- WMMA fragment loaders (ISA 7.12.2 layouts, wave32) ------------------- */
__device__ __forceinline__ v16h frag_a16(const _Float16* base, int ld,
                                         int row0, int k0, int lane) {
  int m = lane & 15, hf = lane >> 4;
  const _Float16* p = base + (row0 + m) * ld + k0 + 8 * hf;
  union { v16h v; v8h h[2]; } u;
  u.h[0] = *(const v8h*)(p);
  u.h[1] = *(const v8h*)(p + 16);
  return u.v;
}

__device__ __forceinline__ v16h frag_b16(const _Float16* base, int ld,
                                         int col0, int k0, int lane) {
  int n = lane & 15, kh = lane >> 4;
  const _Float16* p = base + (col0 + n) * ld + k0 + 8 * kh;
  union { v16h v; v8h h[2]; } u;
  u.h[0] = *(const v8h*)(p);
  u.h[1] = *(const v8h*)(p + 16);
  return u.v;
}

__device__ __forceinline__ v4h cvt4(float4 f) {
  v4h h = { (_Float16)f.x, (_Float16)f.y, (_Float16)f.z, (_Float16)f.w };
  return h;
}

/* ---- Kernel 1: QKV projection GEMM (M=200704, K=384, Ncols=1152) ----------
 * Double-buffered LDS pipeline: iteration i issues global loads of tile i+1,
 * runs WMMAs on tile i, then converts/stores tile i+1 -> one barrier/step.
 */
__global__ __launch_bounds__(256)
void qkv_gemm_kernel(const float* __restrict__ x, const float* __restrict__ w,
                     const float* __restrict__ bias,
                     _Float16* __restrict__ Q, _Float16* __restrict__ K,
                     _Float16* __restrict__ V)
{
  __shared__ _Float16 As[2][64 * 40];
  __shared__ _Float16 Bs[2][64 * 40];
  const int tid  = threadIdx.x;
  const int lane = tid & 31;
  const int wv   = tid >> 5;
  const int mt   = wv & 3;
  const int nt0  = (wv >> 2) * 2;

  const int rowBase = blockIdx.x * 64;
  const int colBase = blockIdx.y * 64;

  /* per-thread staging coordinates (2 float4 chunks each for A and B) */
  const int r0 = tid >> 3,          c0 = (tid & 7) << 2;
  const int r1 = (tid + 256) >> 3,  c1 = c0;
  const float* xa = x + (size_t)(rowBase + r0) * DIM + c0;
  const float* xb = x + (size_t)(rowBase + r1) * DIM + c1;
  const float* wa = w + (size_t)(colBase + r0) * DIM + c0;
  const float* wb = w + (size_t)(colBase + r1) * DIM + c1;

  v8f acc0 = {}; v8f acc1 = {};

  /* prologue: stage tile 0 into buffer 0 */
  {
    float4 fa0 = *(const float4*)xa, fa1 = *(const float4*)xb;
    float4 fb0 = *(const float4*)wa, fb1 = *(const float4*)wb;
    *(v4h*)&As[0][r0 * 40 + c0] = cvt4(fa0);
    *(v4h*)&As[0][r1 * 40 + c1] = cvt4(fa1);
    *(v4h*)&Bs[0][r0 * 40 + c0] = cvt4(fb0);
    *(v4h*)&Bs[0][r1 * 40 + c1] = cvt4(fb1);
  }
  __syncthreads();

  for (int it = 0; it < KSTEPS; ++it) {
    const int buf = it & 1, nbuf = buf ^ 1;
    const bool more = (it + 1) < KSTEPS;
    float4 fa0, fa1, fb0, fb1;
    if (more) {                         /* issue next-tile loads early */
      int kk = (it + 1) * 32;
      fa0 = *(const float4*)(xa + kk);
      fa1 = *(const float4*)(xb + kk);
      fb0 = *(const float4*)(wa + kk);
      fb1 = *(const float4*)(wb + kk);
      if (it + 2 < KSTEPS) {
        __builtin_prefetch(xa + kk + 32, 0, 3);
        __builtin_prefetch(wa + kk + 32, 0, 3);
      }
    }
    v16h a  = frag_a16(As[buf], 40, mt * 16, 0, lane);
    v16h b0 = frag_b16(Bs[buf], 40, nt0 * 16, 0, lane);
    v16h b1 = frag_b16(Bs[buf], 40, (nt0 + 1) * 16, 0, lane);
    acc0 = __builtin_amdgcn_wmma_f32_16x16x32_f16(false, a, false, b0,
                                                  (short)0, acc0, false, false);
    acc1 = __builtin_amdgcn_wmma_f32_16x16x32_f16(false, a, false, b1,
                                                  (short)0, acc1, false, false);
    if (more) {                         /* convert+store behind the WMMAs */
      *(v4h*)&As[nbuf][r0 * 40 + c0] = cvt4(fa0);
      *(v4h*)&As[nbuf][r1 * 40 + c1] = cvt4(fa1);
      *(v4h*)&Bs[nbuf][r0 * 40 + c0] = cvt4(fb0);
      *(v4h*)&Bs[nbuf][r1 * 40 + c1] = cvt4(fb1);
    }
    __syncthreads();
  }

  for (int t = 0; t < 2; ++t) {
    v8f acc = t ? acc1 : acc0;
    int nt  = nt0 + t;
    int col = colBase + nt * 16 + (lane & 15);
    int part = col / QKD;               /* 0=q 1=k 2=v */
    int h = (col % QKD) >> 5;
    int d = col & 31;
    float bb = bias[col];
    for (int r = 0; r < 8; ++r) {
      int row = rowBase + mt * 16 + ((lane >> 4) << 3) + r;
      float val = acc[r] + bb;
      int bw = row / NTOK, n = row % NTOK;
      size_t o = (((size_t)bw * HEADS + h) * NTOK + n) * HD + d;
      if (part == 0)      Q[o] = (_Float16)(val * SCALE);
      else if (part == 1) K[o] = (_Float16)val;
      else                V[o] = (_Float16)val;
    }
  }
}

/* ---- Kernel 2: fused windowed attention per (window, head) ---------------- */
__global__ __launch_bounds__(128)
void attn_kernel(const _Float16* __restrict__ Q, const _Float16* __restrict__ Kt,
                 const _Float16* __restrict__ V,
                 const float* __restrict__ bias_table,
                 const float* __restrict__ mask, _Float16* __restrict__ AO)
{
  __shared__ _Float16 Qs[64 * 40];
  __shared__ _Float16 Ks[64 * 40];
  __shared__ _Float16 Vt[32 * 72];      /* V transposed: [hd][token] */
  __shared__ float    Sm[64 * 64];
  __shared__ _Float16 Pm[64 * 72];

  const int b = blockIdx.x, h = blockIdx.y;
  const int tid = threadIdx.x, lane = tid & 31, wv = tid >> 5;
  const size_t base = (((size_t)b * HEADS + h) * NTOK) * HD;
  const v8f vz = {};
  const v8h hz = {};

  /* stage Q/K via async b128 copies; V via b128 load + transposed scatter */
  for (int iter = 0; iter < 2; ++iter) {
    int idx = tid + iter * 128;                /* 0..255 */
    int t = idx >> 2, c8 = (idx & 3) << 3;
    if (t < NTOK) {
      async_copy_b128(Q  + base + t * HD + c8, &Qs[t * 40 + c8]);
      async_copy_b128(Kt + base + t * HD + c8, &Ks[t * 40 + c8]);
      v8h hv = *(const v8h*)(V + base + t * HD + c8);
      for (int e = 0; e < 8; ++e) Vt[(c8 + e) * 72 + t] = hv[e];
    } else {
      *(v8h*)&Qs[t * 40 + c8] = hz;
      *(v8h*)&Ks[t * 40 + c8] = hz;
      for (int e = 0; e < 8; ++e) Vt[(c8 + e) * 72 + t] = (_Float16)0.f;
    }
  }
  async_wait();
  __syncthreads();

  /* S = Q K^T : K-dim = HD = 32 -> one WMMA per 16x16 tile */
  v8f sacc[4];
  for (int i = 0; i < 4; ++i) sacc[i] = vz;
  {
    v16h a = frag_a16(Qs, 40, wv * 16, 0, lane);
    for (int nt = 0; nt < 4; ++nt) {
      v16h bf = frag_b16(Ks, 40, nt * 16, 0, lane);
      sacc[nt] = __builtin_amdgcn_wmma_f32_16x16x32_f16(false, a, false, bf,
                                                        (short)0, sacc[nt],
                                                        false, false);
    }
  }
  for (int nt = 0; nt < 4; ++nt)
    for (int r = 0; r < 8; ++r) {
      int rr = wv * 16 + ((lane >> 4) << 3) + r;
      int cc = nt * 16 + (lane & 15);
      Sm[rr * 64 + cc] = sacc[nt][r];
    }
  __syncthreads();

  /* bias + mask + row softmax (threads 0..63, one row each) */
  if (tid < 64) {
    int row = tid;
    if (row < NTOK) {
      int ih = row / 7, iw = row % 7;
      const float* mrow = mask + ((size_t)(b % NW) * NTOK + row) * NTOK;
      float mx = -1e30f;
      for (int j = 0; j < NTOK; ++j) {
        int jh = j / 7, jw = j % 7;
        int ridx = (ih - jh + 6) * 13 + (iw - jw + 6);
        float val = Sm[row * 64 + j] + bias_table[ridx * HEADS + h] + mrow[j];
        Sm[row * 64 + j] = val;
        mx = fmaxf(mx, val);
      }
      float ssum = 0.0f;
      for (int j = 0; j < NTOK; ++j) {
        float e = __expf(Sm[row * 64 + j] - mx);
        Sm[row * 64 + j] = e;
        ssum += e;
      }
      float inv = 1.0f / ssum;
      for (int j = 0; j < NTOK; ++j)
        Pm[row * 72 + j] = (_Float16)(Sm[row * 64 + j] * inv);
      for (int j = NTOK; j < 64; ++j) Pm[row * 72 + j] = (_Float16)0.f;
    } else {
      for (int j = 0; j < 64; ++j) Pm[row * 72 + j] = (_Float16)0.f;
    }
  }
  __syncthreads();

  /* O = P V : K-dim = 64 tokens -> 2 WMMA steps; 2 hd tiles */
  v8f oacc[2]; oacc[0] = vz; oacc[1] = vz;
  for (int kk = 0; kk < 2; ++kk) {
    v16h a = frag_a16(Pm, 72, wv * 16, kk * 32, lane);
    for (int nt = 0; nt < 2; ++nt) {
      v16h bf = frag_b16(Vt, 72, nt * 16, kk * 32, lane);
      oacc[nt] = __builtin_amdgcn_wmma_f32_16x16x32_f16(false, a, false, bf,
                                                        (short)0, oacc[nt],
                                                        false, false);
    }
  }
  for (int nt = 0; nt < 2; ++nt) {
    int d = nt * 16 + (lane & 15);
    for (int r = 0; r < 8; ++r) {
      int t = wv * 16 + ((lane >> 4) << 3) + r;
      if (t < NTOK)
        AO[((size_t)b * NTOK + t) * QKD + h * HD + d] = (_Float16)oacc[nt][r];
    }
  }
}

/* ---- Kernel 3: output projection (M=200704, K=384, N=384), f32 out --------
 * Double-buffered; f16 A tile streams via async global->LDS DMA into the back
 * buffer while WMMAs consume the front buffer.
 */
__global__ __launch_bounds__(256)
void proj_gemm_kernel(const _Float16* __restrict__ A, const float* __restrict__ w,
                      const float* __restrict__ bias, float* __restrict__ out)
{
  __shared__ _Float16 As[2][64 * 40];
  __shared__ _Float16 Bs[2][64 * 40];
  const int tid  = threadIdx.x;
  const int lane = tid & 31;
  const int wv   = tid >> 5;
  const int mt   = wv & 3;
  const int nt0  = (wv >> 2) * 2;

  const int rowBase = blockIdx.x * 64;
  const int colBase = blockIdx.y * 64;

  const int ra = tid >> 2,          ca = (tid & 3) << 3;   /* A: 1 v8h chunk */
  const int r0 = tid >> 3,          c0 = (tid & 7) << 2;   /* B: 2 float4     */
  const int r1 = (tid + 256) >> 3,  c1 = c0;
  const _Float16* Aa = A + (size_t)(rowBase + ra) * QKD + ca;
  const float*    wa = w + (size_t)(colBase + r0) * QKD + c0;
  const float*    wb = w + (size_t)(colBase + r1) * QKD + c1;

  v8f acc0 = {}; v8f acc1 = {};

  /* prologue: tile 0 */
  {
    async_copy_b128(Aa, &As[0][ra * 40 + ca]);
    float4 fb0 = *(const float4*)wa, fb1 = *(const float4*)wb;
    *(v4h*)&Bs[0][r0 * 40 + c0] = cvt4(fb0);
    *(v4h*)&Bs[0][r1 * 40 + c1] = cvt4(fb1);
    async_wait();
  }
  __syncthreads();

  for (int it = 0; it < KSTEPS; ++it) {
    const int buf = it & 1, nbuf = buf ^ 1;
    const bool more = (it + 1) < KSTEPS;
    float4 fb0, fb1;
    if (more) {
      int kk = (it + 1) * 32;
      async_copy_b128(Aa + kk, &As[nbuf][ra * 40 + ca]);  /* DMA overlaps WMMA */
      fb0 = *(const float4*)(wa + kk);
      fb1 = *(const float4*)(wb + kk);
      if (it + 2 < KSTEPS) __builtin_prefetch(wa + kk + 32, 0, 3);
    }
    v16h a  = frag_a16(As[buf], 40, mt * 16, 0, lane);
    v16h b0 = frag_b16(Bs[buf], 40, nt0 * 16, 0, lane);
    v16h b1 = frag_b16(Bs[buf], 40, (nt0 + 1) * 16, 0, lane);
    acc0 = __builtin_amdgcn_wmma_f32_16x16x32_f16(false, a, false, b0,
                                                  (short)0, acc0, false, false);
    acc1 = __builtin_amdgcn_wmma_f32_16x16x32_f16(false, a, false, b1,
                                                  (short)0, acc1, false, false);
    if (more) {
      *(v4h*)&Bs[nbuf][r0 * 40 + c0] = cvt4(fb0);
      *(v4h*)&Bs[nbuf][r1 * 40 + c1] = cvt4(fb1);
      async_wait();
    }
    __syncthreads();
  }

  for (int t = 0; t < 2; ++t) {
    v8f acc = t ? acc1 : acc0;
    int nt  = nt0 + t;
    int col = colBase + nt * 16 + (lane & 15);
    float bb = bias[col];
    for (int r = 0; r < 8; ++r) {
      int row = rowBase + mt * 16 + ((lane >> 4) << 3) + r;
      out[(size_t)row * DIM + col] = acc[r] + bb;
    }
  }
}

extern "C" void kernel_launch(void* const* d_in, const int* in_sizes, int n_in,
                              void* d_out, int out_size, void* d_ws, size_t ws_size,
                              hipStream_t stream) {
  const float* x      = (const float*)d_in[0];
  const float* mask   = (const float*)d_in[1];
  const float* w_qkv  = (const float*)d_in[2];
  const float* b_qkv  = (const float*)d_in[3];
  const float* btab   = (const float*)d_in[4];
  const float* w_proj = (const float*)d_in[5];
  const float* b_proj = (const float*)d_in[6];
  float* out = (float*)d_out;

  const size_t qkv_elems = (size_t)B_TOT * HEADS * NTOK * HD;  /* 77,070,336 */
  _Float16* Q  = (_Float16*)d_ws;
  _Float16* K  = Q + qkv_elems;
  _Float16* V  = K + qkv_elems;
  _Float16* AO = V + qkv_elems;   /* [MTOT, QKD] f16, head-major within QKD */

  dim3 g1(MTOT / 64, (3 * QKD) / 64);               /* 3136 x 18 */
  qkv_gemm_kernel<<<g1, 256, 0, stream>>>(x, w_qkv, b_qkv, Q, K, V);

  dim3 g2(B_TOT, HEADS);                            /* 4096 x 12 */
  attn_kernel<<<g2, 128, 0, stream>>>(Q, K, V, btab, mask, AO);

  dim3 g3(MTOT / 64, DIM / 64);                     /* 3136 x 6 */
  proj_gemm_kernel<<<g3, 256, 0, stream>>>(AO, w_proj, b_proj, out);
}
